// Attention_73306501808299
// MI455X (gfx1250) — matrix operand order repaired and checked by reading, not compiled, not run on hardware
//
#include <hip/hip_runtime.h>
#include <hip/hip_bf16.h>
#include <cmath>

// ---------------------------------------------------------------------------
// Problem constants (from the reference)
// ---------------------------------------------------------------------------
#define Bsz   4
#define Lseq  1024
#define Hq    32
#define KVH   8
#define Dh    128
#define DIM   4096
#define Tn    (Bsz * Lseq)            // 4096
#define QKV_OUT ((Hq + 2 * KVH) * Dh) // 6144

typedef __attribute__((ext_vector_type(16))) _Float16 v16h;
typedef __attribute__((ext_vector_type(8)))  float    v8f;

// ---------------------------------------------------------------------------
// WMMA fragment loaders (CDNA5 16-bit layouts, wave32)
//  A-matrix 16x32: lane<16 holds K=0..7 (VGPR0-3), K=16..23 (VGPR4-7)
//                  lane>=16 holds K=8..15, K=24..31
//  B-matrix 32x16: lane<16 (col n) holds K=0..7, K=16..23; lane>=16 holds K=8..15, K=24..31
// ---------------------------------------------------------------------------
__device__ __forceinline__ v16h load_fragA(const _Float16* rowbase, int half) {
  union { uint4 q[2]; v16h v; } u;
  u.q[0] = *(const uint4*)(rowbase + half * 8);
  u.q[1] = *(const uint4*)(rowbase + 16 + half * 8);
  return u.v;
}
__device__ __forceinline__ v16h load_fragB(const _Float16* colbase, int half) {
  union { uint4 q[2]; v16h v; } u;
  u.q[0] = *(const uint4*)(colbase + half * 8);
  u.q[1] = *(const uint4*)(colbase + 16 + half * 8);
  return u.v;
}

// ---------------------------------------------------------------------------
// CDNA5 async global->LDS copy (ASYNCcnt-tracked, no VGPR round trip).
// vdst VGPR holds the per-lane LDS byte offset (low 32 bits of the generic
// shared-memory address, per the flat-aperture truncation rule).
// ---------------------------------------------------------------------------
__device__ __forceinline__ void async_copy16(const _Float16* g, _Float16* lds) {
  unsigned loff = (unsigned)(size_t)lds;
  unsigned long long ga = (unsigned long long)(size_t)g;
  asm volatile("global_load_async_to_lds_b128 %0, %1, off"
               :: "v"(loff), "v"(ga) : "memory");
}
__device__ __forceinline__ void wait_async0() {
  asm volatile("s_wait_asynccnt 0" ::: "memory");
}

// ---------------------------------------------------------------------------
// Kernel 0: bulk f32 -> f16 conversion (one cheap bandwidth pass; makes the
// compute-bound GEMMs pure-f16 so staging needs no VALU conversion work).
// ---------------------------------------------------------------------------
__global__ __launch_bounds__(256)
void cvt_f32_f16(const float* __restrict__ in, _Float16* __restrict__ out,
                 size_t n) {
  size_t i = ((size_t)blockIdx.x * 256 + threadIdx.x) * 8;
  if (i >= n) return;
  float4 a = *(const float4*)&in[i];
  float4 b = *(const float4*)&in[i + 4];
  union { _Float16 h[8]; uint4 q; } u;
  u.h[0] = (_Float16)a.x; u.h[1] = (_Float16)a.y;
  u.h[2] = (_Float16)a.z; u.h[3] = (_Float16)a.w;
  u.h[4] = (_Float16)b.x; u.h[5] = (_Float16)b.y;
  u.h[6] = (_Float16)b.z; u.h[7] = (_Float16)b.w;
  *(uint4*)&out[i] = u.q;
}

// ---------------------------------------------------------------------------
// Kernels 1/4: C[M,N] = A[M,K] * B[N,K]^T (+bias), all-f16 operands.
// Block tile 128x128, K-step 32, 256 threads = 8 waves (wave32), each wave a
// 64x32 sub-tile (4 M-frags x 2 N-frags).  Double-buffered LDS staging via
// global_load_async_to_lds_b128 with s_wait_asynccnt pipelining.
// ---------------------------------------------------------------------------
template <bool OUT_F32>
__global__ __launch_bounds__(256)
void gemm_nt_wmma(const _Float16* __restrict__ A, const _Float16* __restrict__ Bw,
                  const float* __restrict__ bias, void* __restrict__ Cout,
                  int M, int N, int K) {
  __shared__ __align__(16) _Float16 As[2][128 * 32];
  __shared__ __align__(16) _Float16 Bs[2][128 * 32];

  const int tid  = threadIdx.x;
  const int lane = tid & 31;
  const int wave = tid >> 5;
  const int wm   = wave >> 2;   // 0..1 -> M offset wm*64
  const int wn   = wave & 3;    // 0..3 -> N offset wn*32
  const int half = lane >> 4;
  const int l    = lane & 15;
  const int m0   = blockIdx.y * 128;
  const int n0   = blockIdx.x * 128;

  v8f acc[4][2] = {};

  // Stage one 128x32 f16 tile of A and B into LDS buffer `buf` (async).
  auto stage = [&](int k0, int buf) {
    #pragma unroll
    for (int i = 0; i < 2; ++i) {
      int c   = tid + 256 * i;       // 512 16-byte chunks per matrix
      int row = c >> 2, c8 = (c & 3) * 8;
      async_copy16(&A[(size_t)(m0 + row) * K + k0 + c8], &As[buf][row * 32 + c8]);
      async_copy16(&Bw[(size_t)(n0 + row) * K + k0 + c8], &Bs[buf][row * 32 + c8]);
    }
  };

  stage(0, 0);
  wait_async0();
  __syncthreads();

  const int nk = K >> 5;
  for (int kt = 0; kt < nk; ++kt) {
    const int cur = kt & 1;
    if (kt + 1 < nk) stage((kt + 1) << 5, cur ^ 1);   // prefetch next tile

    v16h bfrag[2];
    #pragma unroll
    for (int nf = 0; nf < 2; ++nf)
      bfrag[nf] = load_fragB(&Bs[cur][(wn * 32 + nf * 16 + l) * 32], half);
    #pragma unroll
    for (int mf = 0; mf < 4; ++mf) {
      v16h a = load_fragA(&As[cur][(wm * 64 + mf * 16 + l) * 32], half);
      #pragma unroll
      for (int nf = 0; nf < 2; ++nf)
        acc[mf][nf] = __builtin_amdgcn_wmma_f32_16x16x32_f16(
            false, a, false, bfrag[nf], (short)0, acc[mf][nf], false, false);
    }

    wait_async0();     // prefetch complete
    __syncthreads();   // all waves done reading `cur` before it is re-staged
  }

  // --- epilogue: C layout row = r + 8*half, col = lane&15 ---
  #pragma unroll
  for (int mf = 0; mf < 4; ++mf)
    #pragma unroll
    for (int nf = 0; nf < 2; ++nf) {
      int col = n0 + wn * 32 + nf * 16 + l;
      float bv = bias ? bias[col] : 0.0f;
      #pragma unroll
      for (int r = 0; r < 8; ++r) {
        int row = m0 + wm * 64 + mf * 16 + r + 8 * half;
        float v = acc[mf][nf][r] + bv;
        if constexpr (OUT_F32)
          ((float*)Cout)[(size_t)row * N + col] = v;
        else
          ((_Float16*)Cout)[(size_t)row * N + col] = (_Float16)v;
      }
    }
}

// ---------------------------------------------------------------------------
// Kernel 2: RoPE + scatter.  Reads xqkv[t][6144] (f16), writes:
//   Q  [t][h][d]            (t-major, h<32, rotated)
//   K  [b][kvh][s][d]       (rotated)
//   Vt [b][kvh][d][s]       (TRANSPOSED so P*V B-fragments are contiguous)
// ---------------------------------------------------------------------------
__global__ __launch_bounds__(256)
void rope_scatter(const _Float16* __restrict__ xqkv,
                  const float* __restrict__ cosT, const float* __restrict__ sinT,
                  _Float16* __restrict__ Qb, _Float16* __restrict__ Kb,
                  _Float16* __restrict__ Vt) {
  int id = blockIdx.x * 256 + threadIdx.x;       // T * 48 * 64 threads
  int dp = id & 63;
  int rest = id >> 6;
  int hh = rest % (Hq + 2 * KVH);
  int t  = rest / (Hq + 2 * KVH);
  if (t >= Tn) return;

  float x1 = (float)xqkv[(size_t)t * QKV_OUT + hh * Dh + dp];
  float x2 = (float)xqkv[(size_t)t * QKV_OUT + hh * Dh + dp + 64];
  float o1, o2;
  if (hh < Hq + KVH) {
    float c = cosT[t * 64 + dp], s = sinT[t * 64 + dp];
    o1 = x1 * c - x2 * s;
    o2 = x2 * c + x1 * s;
  } else {
    o1 = x1; o2 = x2;
  }
  int b = t >> 10, sidx = t & (Lseq - 1);
  if (hh < Hq) {
    _Float16* q = &Qb[(size_t)t * (Hq * Dh) + hh * Dh];
    q[dp] = (_Float16)o1; q[dp + 64] = (_Float16)o2;
  } else if (hh < Hq + KVH) {
    _Float16* k = &Kb[(((size_t)b * KVH + (hh - Hq)) * Lseq + sidx) * Dh];
    k[dp] = (_Float16)o1; k[dp + 64] = (_Float16)o2;
  } else {
    _Float16* v = &Vt[(((size_t)b * KVH + (hh - Hq - KVH)) * Dh) * Lseq];
    v[(size_t)dp * Lseq + sidx]        = (_Float16)o1;
    v[(size_t)(dp + 64) * Lseq + sidx] = (_Float16)o2;
  }
}

// ---------------------------------------------------------------------------
// Kernel 3: causal GQA flash attention.
// grid = (L/64, H, B); 128 threads = 4 waves; wave w owns 16 query rows.
// S = Q*K^T via WMMA (K B-frags straight from global), online softmax on the
// WMMA C layout, P routed through per-wave LDS to become an A-fragment,
// O += P*V via WMMA (V^T B-frags straight from global).
// ---------------------------------------------------------------------------
__global__ __launch_bounds__(128)
void attn_fwd(const _Float16* __restrict__ Qb, const _Float16* __restrict__ Kb,
              const _Float16* __restrict__ Vt, _Float16* __restrict__ Ob) {
  __shared__ __align__(16) _Float16 Plds[4][16][64];

  const int lane = threadIdx.x & 31;
  const int wave = threadIdx.x >> 5;
  const int half = lane >> 4;
  const int l    = lane & 15;
  const int mt   = blockIdx.x;
  const int h    = blockIdx.y;
  const int b    = blockIdx.z;
  const int kvh  = h >> 2;                       // H/KVH = 4
  const int q0   = mt * 64 + wave * 16;
  const float scale = 0.08838834764831845f;      // 1/sqrt(128)

  const size_t kvBase = ((size_t)b * KVH + kvh);

  // Q A-fragments for the whole loop (4 K-chunks over D=128)
  v16h qf[4];
  #pragma unroll
  for (int kf = 0; kf < 4; ++kf)
    qf[kf] = load_fragA(
        &Qb[((size_t)(b * Lseq) + q0 + l) * (Hq * Dh) + h * Dh + kf * 32], half);

  float mrow[8], lrow[8];
  v8f   o[8] = {};
  #pragma unroll
  for (int r = 0; r < 8; ++r) { mrow[r] = -__builtin_inff(); lrow[r] = 0.0f; }

  const int jEnd = q0 + 15;                      // per-wave causal bound
  for (int j0 = 0; j0 <= jEnd; j0 += 64) {
    // ---- S = Q * K^T (4 N-frags of 16 keys) ----
    v8f s[4] = {};
    #pragma unroll
    for (int nf = 0; nf < 4; ++nf) {
      const _Float16* krow = &Kb[(kvBase * Lseq + (j0 + nf * 16 + l)) * Dh];
      #pragma unroll
      for (int kf = 0; kf < 4; ++kf) {
        v16h kfrag = load_fragB(krow + kf * 32, half);
        s[nf] = __builtin_amdgcn_wmma_f32_16x16x32_f16(
            false, qf[kf], false, kfrag, (short)0, s[nf], false, false);
      }
    }
    // ---- scale + causal mask + row max ----
    float rmax[8];
    #pragma unroll
    for (int r = 0; r < 8; ++r) rmax[r] = -__builtin_inff();
    #pragma unroll
    for (int nf = 0; nf < 4; ++nf) {
      int n = j0 + nf * 16 + l;
      #pragma unroll
      for (int r = 0; r < 8; ++r) {
        int m = q0 + r + 8 * half;
        float v = s[nf][r] * scale;
        if (n > m) v = -__builtin_inff();
        s[nf][r] = v;
        rmax[r] = fmaxf(rmax[r], v);
      }
    }
    #pragma unroll
    for (int off = 1; off < 16; off <<= 1)
      #pragma unroll
      for (int r = 0; r < 8; ++r)
        rmax[r] = fmaxf(rmax[r], __shfl_xor(rmax[r], off, 32));

    // ---- online softmax update ----
    float corr[8], rsum[8];
    #pragma unroll
    for (int r = 0; r < 8; ++r) {
      float mnew = fmaxf(mrow[r], rmax[r]);
      corr[r] = __expf(mrow[r] - mnew);
      mrow[r] = mnew;
      rsum[r] = 0.0f;
    }
    #pragma unroll
    for (int nf = 0; nf < 4; ++nf)
      #pragma unroll
      for (int r = 0; r < 8; ++r) {
        float p = __expf(s[nf][r] - mrow[r]);
        s[nf][r] = p;
        rsum[r] += p;
      }
    #pragma unroll
    for (int off = 1; off < 16; off <<= 1)
      #pragma unroll
      for (int r = 0; r < 8; ++r)
        rsum[r] += __shfl_xor(rsum[r], off, 32);
    #pragma unroll
    for (int r = 0; r < 8; ++r) lrow[r] = lrow[r] * corr[r] + rsum[r];
    #pragma unroll
    for (int df = 0; df < 8; ++df)
      #pragma unroll
      for (int r = 0; r < 8; ++r) o[df][r] *= corr[r];

    // ---- C-layout P -> A-fragment via per-wave LDS (same-wave, in-order) ----
    #pragma unroll
    for (int nf = 0; nf < 4; ++nf)
      #pragma unroll
      for (int r = 0; r < 8; ++r)
        Plds[wave][r + 8 * half][nf * 16 + l] = (_Float16)s[nf][r];
    __asm__ volatile("s_wait_dscnt 0" ::: "memory");

    v16h pf[2];
    #pragma unroll
    for (int kf2 = 0; kf2 < 2; ++kf2)
      pf[kf2] = load_fragA(&Plds[wave][l][kf2 * 32], half);

    // ---- O += P * V  (V^T rows are key-contiguous: direct B-frags) ----
    #pragma unroll
    for (int df = 0; df < 8; ++df) {
      const _Float16* vrow = &Vt[(kvBase * Dh + df * 16 + l) * Lseq + j0];
      #pragma unroll
      for (int kf2 = 0; kf2 < 2; ++kf2) {
        v16h vf = load_fragB(vrow + kf2 * 32, half);
        o[df] = __builtin_amdgcn_wmma_f32_16x16x32_f16(
            false, pf[kf2], false, vf, (short)0, o[df], false, false);
      }
    }
  }

  // ---- normalize + store f16 [t][h*128+d] ----
  #pragma unroll
  for (int df = 0; df < 8; ++df)
    #pragma unroll
    for (int r = 0; r < 8; ++r) {
      int m = q0 + r + 8 * half;
      size_t t = (size_t)b * Lseq + m;
      Ob[t * (Hq * Dh) + h * Dh + df * 16 + l] =
          (_Float16)(o[df][r] / lrow[r]);
    }
}

// ---------------------------------------------------------------------------
// Launch:
//   cvt(x,w_qkv,w_o) -> QKV GEMM -> RoPE scatter -> flash attn -> O GEMM
// Workspace (~250 MB): f16 copies of x/weights + xqkv, Q, K, V^T, attnOut.
// ---------------------------------------------------------------------------
extern "C" void kernel_launch(void* const* d_in, const int* in_sizes, int n_in,
                              void* d_out, int out_size, void* d_ws, size_t ws_size,
                              hipStream_t stream) {
  const float* x     = (const float*)d_in[0];
  const float* cosT  = (const float*)d_in[1];
  const float* sinT  = (const float*)d_in[2];
  const float* qkv_w = (const float*)d_in[3];
  const float* qkv_b = (const float*)d_in[4];
  const float* o_w   = (const float*)d_in[5];
  // d_in[6..11] (caches/indices) not needed: caches start zeroed and the
  // output only depends on the freshly written K/V.

  size_t off = 0;
  auto alloc = [&](size_t bytes) -> void* {
    void* p = (char*)d_ws + off;
    off += (bytes + 255) & ~(size_t)255;
    return p;
  };
  _Float16* x16   = (_Float16*)alloc((size_t)Tn * DIM * 2);           // 33.6 MB
  _Float16* qkvw16= (_Float16*)alloc((size_t)QKV_OUT * DIM * 2);      // 50.3 MB
  _Float16* ow16  = (_Float16*)alloc((size_t)DIM * DIM * 2);          // 33.6 MB
  _Float16* xqkv  = (_Float16*)alloc((size_t)Tn * QKV_OUT * 2);       // 50.3 MB
  _Float16* Qb    = (_Float16*)alloc((size_t)Tn * Hq * Dh * 2);       // 33.6 MB
  _Float16* Kb    = (_Float16*)alloc((size_t)Bsz * KVH * Lseq * Dh * 2); // 8.4 MB
  _Float16* Vt    = (_Float16*)alloc((size_t)Bsz * KVH * Dh * Lseq * 2); // 8.4 MB
  _Float16* AO    = (_Float16*)alloc((size_t)Tn * Hq * Dh * 2);       // 33.6 MB
  (void)ws_size; (void)in_sizes; (void)n_in; (void)out_size;

  // 0) f32 -> f16 conversions (activations + weights)
  {
    size_t nx = (size_t)Tn * DIM, nq = (size_t)QKV_OUT * DIM, no = (size_t)DIM * DIM;
    cvt_f32_f16<<<(unsigned)((nx / 8 + 255) / 256), 256, 0, stream>>>(x, x16, nx);
    cvt_f32_f16<<<(unsigned)((nq / 8 + 255) / 256), 256, 0, stream>>>(qkv_w, qkvw16, nq);
    cvt_f32_f16<<<(unsigned)((no / 8 + 255) / 256), 256, 0, stream>>>(o_w, ow16, no);
  }

  // 1) xqkv = x @ qkv_w^T + qkv_b   (4096 x 6144 x 4096)
  gemm_nt_wmma<false><<<dim3(QKV_OUT / 128, Tn / 128), 256, 0, stream>>>(
      x16, qkvw16, qkv_b, xqkv, Tn, QKV_OUT, DIM);

  // 2) RoPE + scatter to Q / K / V^T
  rope_scatter<<<(Tn * (Hq + 2 * KVH) * 64) / 256, 256, 0, stream>>>(
      xqkv, cosT, sinT, Qb, Kb, Vt);

  // 3) causal GQA flash attention
  attn_fwd<<<dim3(Lseq / 64, Hq, Bsz), 128, 0, stream>>>(Qb, Kb, Vt, AO);

  // 4) out = attnOut @ o_w^T  (4096 x 4096 x 4096), f32 output
  gemm_nt_wmma<true><<<dim3(DIM / 128, Tn / 128), 256, 0, stream>>>(
      AO, ow16, nullptr, d_out, Tn, DIM, DIM);
}
